// RetNet_18837726560273
// MI455X (gfx1250) — matrix operand order repaired and checked by reading, not compiled, not run on hardware
//
#include <hip/hip_runtime.h>
#include <hip/hip_bf16.h>
#include <math.h>
#include <stdint.h>

typedef _Float16 half8 __attribute__((ext_vector_type(8)));
typedef _Float16 v16h  __attribute__((ext_vector_type(16)));
typedef float    v8f   __attribute__((ext_vector_type(8)));

#define SEQL   2048
#define BATCH  2
#define HID    128
#define NHEADS 4
#define HEADD  32
#define HEADV  64
#define VD     256
#define FFND   256
#define ROWS   (BATCH*SEQL)   // 4096

static __device__ __forceinline__ v16h mk16(half8 lo, half8 hi) {
  return __builtin_shufflevector(lo, hi, 0,1,2,3,4,5,6,7,8,9,10,11,12,13,14,15);
}
static __device__ __forceinline__ v8f wmma16(v16h a, v16h b, v8f c) {
  // D = A(16x32 f16) * B(32x16 f16) + C(16x16 f32)
  return __builtin_amdgcn_wmma_f32_16x16x32_f16(false, a, false, b, (short)0, c, false, false);
}
static __device__ __forceinline__ float wave_sum(float v) {
#pragma unroll
  for (int m = 16; m > 0; m >>= 1) v += __shfl_xor(v, m, 32);
  return v;
}
// async copy 16B global -> LDS (per-lane addresses), tracked by ASYNCcnt
static __device__ __forceinline__ void async_g2l_b128(const void* gsrc, void* ldst) {
  uint32_t loff = (uint32_t)(uintptr_t)ldst;   // LDS aperture: low 32 bits = byte offset
  asm volatile("global_load_async_to_lds_b128 %0, %1, off"
               :: "v"(loff), "v"(gsrc) : "memory");
}
static __device__ __forceinline__ void async_wait0() {
  asm volatile("s_wait_asynccnt 0x0" ::: "memory");
}

// ---------- weight repack: f32 KxN row-major -> f16 [n][K] (col-major), per head (grid.z) ----------
__global__ void k_tcast(const float* __restrict__ S, _Float16* __restrict__ D, int K, int N) {
  int idx = blockIdx.x * 256 + threadIdx.x;
  if (idx >= K * N) return;
  int k = idx / N, n = idx % N;
  size_t hb = (size_t)blockIdx.z * K * N;
  D[hb + (size_t)n * K + k] = (_Float16)S[hb + (size_t)k * N + n];
}

// ---------- LayerNorm over HID=128, one wave per row, f16 out ----------
__global__ void k_ln(const float* __restrict__ X, const float* __restrict__ g,
                     const float* __restrict__ b, _Float16* __restrict__ O) {
  int row  = blockIdx.x * 4 + (threadIdx.x >> 5);
  int lane = threadIdx.x & 31;
  const float* x = X + (size_t)row * HID;
  float v[4], s = 0.f, s2 = 0.f;
#pragma unroll
  for (int i = 0; i < 4; ++i) { v[i] = x[lane + 32*i]; s += v[i]; s2 += v[i]*v[i]; }
  s = wave_sum(s); s2 = wave_sum(s2);
  float mu = s * (1.f/HID);
  float inv = rsqrtf(s2 * (1.f/HID) - mu*mu + 1e-5f);
#pragma unroll
  for (int i = 0; i < 4; ++i) {
    int c = lane + 32*i;
    O[(size_t)row*HID + c] = (_Float16)((v[i]-mu)*inv*g[c] + b[c]);
  }
}

// ---------- residual add + LayerNorm: y = R + Xin (stored f32), O = ln(y) f16 ----------
__global__ void k_resid_ln(const float* __restrict__ R, const float* __restrict__ Xin,
                           const float* __restrict__ g, const float* __restrict__ b,
                           float* __restrict__ Yout, _Float16* __restrict__ O) {
  int row  = blockIdx.x * 4 + (threadIdx.x >> 5);
  int lane = threadIdx.x & 31;
  size_t base = (size_t)row * HID;
  float v[4], s = 0.f, s2 = 0.f;
#pragma unroll
  for (int i = 0; i < 4; ++i) {
    int c = lane + 32*i;
    float t = R[base+c] + Xin[base+c];
    Yout[base+c] = t; v[i] = t; s += t; s2 += t*t;
  }
  s = wave_sum(s); s2 = wave_sum(s2);
  float mu = s * (1.f/HID);
  float inv = rsqrtf(s2 * (1.f/HID) - mu*mu + 1e-5f);
#pragma unroll
  for (int i = 0; i < 4; ++i) {
    int c = lane + 32*i;
    O[base+c] = (_Float16)((v[i]-mu)*inv*g[c] + b[c]);
  }
}

// ---------- WMMA GEMM, K templated (128/256), B panel async-staged in LDS ----------
// C(MxN,f32,row) = A(MxK,f16,row) * B(col-major [n][K], f16)
// block = 128 threads (4 waves); each wave computes 32x64; workgroup 128 rows x 64-col panel.
// grid.x = (M/128) * (N/64)
template<int K>
__global__ void k_gemm_t(const _Float16* __restrict__ A, const _Float16* __restrict__ B,
                         float* __restrict__ C, int M, int N) {
  const int nPanels = N >> 6;
  const int pid = blockIdx.x % nPanels;       // 64-col panel
  const int rg  = blockIdx.x / nPanels;       // 128-row group
  const int lane = threadIdx.x & 31, wid = threadIdx.x >> 5;
  const int l16 = lane & 15, hi = lane >> 4;

  __shared__ _Float16 sB[64 * K];             // 16KB (K=128) / 32KB (K=256)
  // async-stage B panel (contiguous 64*K halves) into LDS
  {
    const _Float16* Bp = B + (size_t)pid * 64 * K;
#pragma unroll 4
    for (int i = threadIdx.x; i < (64 * K) / 8; i += 128)
      async_g2l_b128(Bp + (size_t)i * 8, &sB[i * 8]);
    async_wait0();
    __syncthreads();
  }

  const int m0 = rg * 128 + wid * 32;
  v8f acc[2][4];
#pragma unroll
  for (int mf = 0; mf < 2; ++mf)
#pragma unroll
    for (int nf = 0; nf < 4; ++nf)
#pragma unroll
      for (int r = 0; r < 8; ++r) acc[mf][nf][r] = 0.f;

  const _Float16* A0 = A + (size_t)(m0 + l16) * K;
  const _Float16* A1 = A0 + (size_t)16 * K;
#pragma unroll
  for (int k0 = 0; k0 < K; k0 += 32) {
    v16h a0 = mk16(*(const half8*)(A0 + k0 + hi*16), *(const half8*)(A0 + k0 + hi*16 + 8));
    v16h a1 = mk16(*(const half8*)(A1 + k0 + hi*16), *(const half8*)(A1 + k0 + hi*16 + 8));
#pragma unroll
    for (int nf = 0; nf < 4; ++nf) {
      const _Float16* Bc = sB + (nf*16 + l16) * K + k0 + hi*16;
      v16h bb = mk16(*(const half8*)Bc, *(const half8*)(Bc + 8));
      acc[0][nf] = wmma16(a0, bb, acc[0][nf]);
      acc[1][nf] = wmma16(a1, bb, acc[1][nf]);
    }
  }
#pragma unroll
  for (int mf = 0; mf < 2; ++mf)
#pragma unroll
    for (int nf = 0; nf < 4; ++nf)
#pragma unroll
      for (int r = 0; r < 8; ++r)
        C[(size_t)(m0 + mf*16 + r + hi*8) * N + pid*64 + nf*16 + l16] = acc[mf][nf][r];
}

// ---------- xPos rotary: Cbuf (4096x128 f32, cols=h*32+e) -> f16 [b][h][s][e] ----------
__global__ void k_xpos(const float* __restrict__ Qf, _Float16* __restrict__ O, int invert) {
  int idx = blockIdx.x * 256 + threadIdx.x;          // 4096*64 pairs
  int row = idx >> 6, p = idx & 63;
  int h = p >> 4, i = p & 15;
  int b = row >> 11, s = row & (SEQL-1);
  size_t src = (size_t)row * HID + h*HEADD + 2*i;
  float q0 = Qf[src], q1 = Qf[src + 1];
  float sv   = (2.f*i + 12.8f) * (1.f/44.8f);        // (2i + 0.4d)/(1.4d), d=32
  float scl  = __powf(sv, (float)s * (1.f/512.f));
  if (invert) scl = 1.f / scl;
  float freq = __expf(-(float)i * (9.2103403720f/16.f));   // 10000^(-i/16)
  float ang  = (float)s * freq;
  float sn = __sinf(ang) * scl, cs = __cosf(ang) * scl;
  _Float16* o = O + ((size_t)(b*NHEADS + h)*SEQL + s)*HEADD + 2*i;
  o[0] = (_Float16)(q0*cs - q1*sn);
  o[1] = (_Float16)(q1*cs + q0*sn);
}

// ---------- V reshape+cast+TRANSPOSE: Cbuf (4096x256) -> f16 VT [b][h][v][t] ----------
__global__ void k_vcast(const float* __restrict__ C, _Float16* __restrict__ VT) {
  int idx = blockIdx.x * 256 + threadIdx.x;          // 4096*256
  int row = idx >> 8, col = idx & 255;
  int b = row >> 11, s = row & (SEQL-1);
  int h = col >> 6, v = col & 63;
  VT[((size_t)(b*NHEADS + h)*HEADV + v)*SEQL + s] = (_Float16)C[idx];
}

// ---------- fused retention: per (b,h,64-row block): Y = (QK^T .* decay) V ----------
// K tile async-staged in LDS; V read as B operand directly from transposed global VT.
__global__ void k_attn(const _Float16* __restrict__ Q, const _Float16* __restrict__ Kt,
                       const _Float16* __restrict__ VT, float* __restrict__ Y) {
  const int rb = blockIdx.x, h = blockIdx.y, b = blockIdx.z;
  const int lane = threadIdx.x & 31, wid = threadIdx.x >> 5;
  const int l16 = lane & 15, hi = lane >> 4;
  const float gam = 1.0f - __expf(-3.4657359028f - 0.9241962407f * (float)h);
  const float lg  = __logf(gam);
  __shared__ _Float16 sS[64*64];            // 8KB: S tile (C->A layout shuffle)
  __shared__ _Float16 sK[64*HEADD];         // 4KB: K tile
  const size_t base  = (size_t)(b*NHEADS + h) * SEQL;
  const size_t baseV = (size_t)(b*NHEADS + h) * HEADV;
  v8f yacc[4];
#pragma unroll
  for (int nf = 0; nf < 4; ++nf)
#pragma unroll
    for (int r = 0; r < 8; ++r) yacc[nf][r] = 0.f;
  // Q A-fragment for this wave's 16 rows (constant over t loop), lda = 32
  const _Float16* Arow = Q + (base + rb*64 + wid*16 + l16) * HEADD;
  v16h aq = mk16(*(const half8*)(Arow + hi*16), *(const half8*)(Arow + hi*16 + 8));

  for (int t0 = 0; t0 <= rb*64; t0 += 64) {
    // async-stage K tile (contiguous 64*32 halves)
    {
      const _Float16* Kp = Kt + (base + t0) * HEADD;
#pragma unroll
      for (int i = threadIdx.x; i < (64*HEADD)/8; i += 128)
        async_g2l_b128(Kp + (size_t)i * 8, &sK[i * 8]);
      async_wait0();
      __syncthreads();
    }
    // S = Q K^T (B operand from LDS), decay, -> sS (f16)
#pragma unroll
    for (int nt = 0; nt < 4; ++nt) {
      const _Float16* Bc = sK + (nt*16 + l16) * HEADD + hi*16;
      v16h bk = mk16(*(const half8*)Bc, *(const half8*)(Bc + 8));
      v8f s;
#pragma unroll
      for (int r = 0; r < 8; ++r) s[r] = 0.f;
      s = wmma16(aq, bk, s);
#pragma unroll
      for (int r = 0; r < 8; ++r) {
        int srow = rb*64 + wid*16 + r + hi*8;
        int tcol = t0 + nt*16 + l16;
        int diff = srow - tcol;
        float val = (diff >= 0) ? s[r] * __expf((float)diff * lg) : 0.f;
        sS[(wid*16 + r + hi*8)*64 + nt*16 + l16] = (_Float16)val;
      }
    }
    __syncthreads();
    // Y += S * V  (A from sS; B = 16 contiguous halves per lane from global VT)
#pragma unroll
    for (int kb = 0; kb < 64; kb += 32) {
      const _Float16* Srow = sS + (wid*16 + l16)*64 + kb;
      v16h as = mk16(*(const half8*)(Srow + hi*16), *(const half8*)(Srow + hi*16 + 8));
#pragma unroll
      for (int nv = 0; nv < 4; ++nv) {
        const _Float16* Bc = VT + (baseV + nv*16 + l16)*SEQL + t0 + kb + hi*16;
        v16h bv = mk16(*(const half8*)Bc, *(const half8*)(Bc + 8));
        yacc[nv] = wmma16(as, bv, yacc[nv]);
      }
    }
    __syncthreads();   // protect sS/sK before next iteration overwrites
  }
  // store Y -> [b][s][h*64+v] f32
  float* Yb = Y + ((size_t)b*SEQL + rb*64) * VD + h*HEADV;
#pragma unroll
  for (int nv = 0; nv < 4; ++nv)
#pragma unroll
    for (int r = 0; r < 8; ++r)
      Yb[(size_t)(wid*16 + r + hi*8) * VD + nv*16 + l16] = yacc[nv][r];
}

// ---------- GroupNorm(per-head, 64) + SiLU gate -> f16 A-operand (4096x256) ----------
__global__ void k_gngate(const float* __restrict__ Y, const float* __restrict__ G,
                         const float* __restrict__ gg, const float* __restrict__ gb,
                         _Float16* __restrict__ O) {
  int idx  = blockIdx.x * 4 + (threadIdx.x >> 5);   // row*4 + h
  int row  = idx >> 2, h = idx & 3;
  int lane = threadIdx.x & 31;
  const float* y = Y + (size_t)row * VD + h * HEADV;
  float a = y[lane], c = y[lane + 32];
  float s  = wave_sum(a + c);
  float s2 = wave_sum(a*a + c*c);
  float mu  = s  * (1.f/HEADV);
  float inv = rsqrtf(s2 * (1.f/HEADV) - mu*mu + 1e-5f);
#pragma unroll
  for (int j = 0; j < 2; ++j) {
    int e = lane + 32*j, col = h*HEADV + e;
    float yn = ((j ? c : a) - mu) * inv * gg[col] + gb[col];
    float gv = G[(size_t)row * VD + col];
    float sil = gv / (1.f + __expf(-gv));
    O[(size_t)row * VD + col] = (_Float16)(sil * yn);
  }
}

// ---------- GELU(exact) with bias -> f16 ----------
__global__ void k_gelu(const float* __restrict__ C, const float* __restrict__ b1,
                       _Float16* __restrict__ O) {
  int idx = blockIdx.x * 256 + threadIdx.x;          // 4096*256
  float x = C[idx] + b1[idx & (FFND-1)];
  O[idx] = (_Float16)(0.5f * x * (1.f + erff(x * 0.7071067812f)));
}

// ---------- final add: x_out = C + b2 + y ----------
__global__ void k_addout(const float* __restrict__ C, const float* __restrict__ b2,
                         const float* __restrict__ y, float* __restrict__ X) {
  int idx = blockIdx.x * 256 + threadIdx.x;          // 4096*128
  X[idx] = C[idx] + b2[idx & (HID-1)] + y[idx];
}

extern "C" void kernel_launch(void* const* d_in, const int* in_sizes, int n_in,
                              void* d_out, int out_size, void* d_ws, size_t ws_size,
                              hipStream_t stream) {
  (void)in_sizes; (void)n_in; (void)out_size; (void)ws_size;
  const float* x_in  = (const float*)d_in[0];
  const float* ln1_g = (const float*)d_in[1];
  const float* ln1_b = (const float*)d_in[2];
  const float* ln2_g = (const float*)d_in[3];
  const float* ln2_b = (const float*)d_in[4];
  const float* WQ    = (const float*)d_in[5];
  const float* WK    = (const float*)d_in[6];
  const float* WV    = (const float*)d_in[7];
  const float* WG    = (const float*)d_in[8];
  const float* WO    = (const float*)d_in[9];
  const float* gn_g  = (const float*)d_in[10];
  const float* gn_b  = (const float*)d_in[11];
  const float* w1    = (const float*)d_in[12];
  const float* b1    = (const float*)d_in[13];
  const float* w2    = (const float*)d_in[14];
  const float* b2    = (const float*)d_in[15];

  // workspace carve
  char* ws = (char*)d_ws;
  size_t off = 0;
  auto carve = [&](size_t bytes) -> char* {
    char* p = ws + off; off += (bytes + 255) & ~(size_t)255; return p;
  };
  float* Cbuf = (float*)carve((size_t)ROWS*VD*4);     // generic GEMM output
  float* Gf   = (float*)carve((size_t)ROWS*VD*4);     // gate pre-activation
  float* Ybuf = (float*)carve((size_t)ROWS*VD*4);     // retention output
  float* ybuf = (float*)carve((size_t)ROWS*HID*4);    // attn residual
  float* xbuf = (float*)carve((size_t)ROWS*HID*4);    // inter-layer x
  _Float16* h16 = (_Float16*)carve((size_t)ROWS*HID*2);
  _Float16* Q16 = (_Float16*)carve((size_t)BATCH*NHEADS*SEQL*HEADD*2);
  _Float16* K16 = (_Float16*)carve((size_t)BATCH*NHEADS*SEQL*HEADD*2);
  _Float16* VT16= (_Float16*)carve((size_t)BATCH*NHEADS*SEQL*HEADV*2);
  _Float16* a16 = (_Float16*)carve((size_t)ROWS*VD*2);
  _Float16* wq_h = (_Float16*)carve((size_t)HID*HID*2);
  _Float16* wk_h = (_Float16*)carve((size_t)HID*HID*2);
  _Float16* wv_h = (_Float16*)carve((size_t)HID*VD*2);
  _Float16* wg_h = (_Float16*)carve((size_t)HID*VD*2);
  _Float16* wo_h = (_Float16*)carve((size_t)VD*HID*2);
  _Float16* w1_h = (_Float16*)carve((size_t)HID*FFND*2);
  _Float16* w2_h = (_Float16*)carve((size_t)FFND*HID*2);

  for (int L = 0; L < 2; ++L) {
    const float* xcur = (L == 0) ? x_in : xbuf;
    float* xout = (L == 1) ? (float*)d_out : xbuf;
    // repack weights -> f16 col-major [n][K] (per-head blocks for Q/K/V)
    k_tcast<<<dim3(16,1,NHEADS), 256, 0, stream>>>(WQ + (size_t)L*NHEADS*HID*HEADD, wq_h, HID, HEADD);
    k_tcast<<<dim3(16,1,NHEADS), 256, 0, stream>>>(WK + (size_t)L*NHEADS*HID*HEADD, wk_h, HID, HEADD);
    k_tcast<<<dim3(32,1,NHEADS), 256, 0, stream>>>(WV + (size_t)L*NHEADS*HID*HEADV, wv_h, HID, HEADV);
    k_tcast<<<dim3(128,1,1),     256, 0, stream>>>(WG + (size_t)L*HID*VD,  wg_h, HID, VD);
    k_tcast<<<dim3(128,1,1),     256, 0, stream>>>(WO + (size_t)L*VD*HID,  wo_h, VD,  HID);
    k_tcast<<<dim3(128,1,1),     256, 0, stream>>>(w1 + (size_t)L*HID*FFND, w1_h, HID, FFND);
    k_tcast<<<dim3(128,1,1),     256, 0, stream>>>(w2 + (size_t)L*FFND*HID, w2_h, FFND, HID);
    // h = LN(x)
    k_ln<<<ROWS/4, 128, 0, stream>>>(xcur, ln1_g + L*HID, ln1_b + L*HID, h16);
    // Q/K with xPos
    k_gemm_t<HID><<<(ROWS/128)*(HID/64), 128, 0, stream>>>(h16, wq_h, Cbuf, ROWS, HID);
    k_xpos<<<ROWS*64/256, 256, 0, stream>>>(Cbuf, Q16, 0);
    k_gemm_t<HID><<<(ROWS/128)*(HID/64), 128, 0, stream>>>(h16, wk_h, Cbuf, ROWS, HID);
    k_xpos<<<ROWS*64/256, 256, 0, stream>>>(Cbuf, K16, 1);
    // V (stored transposed per head)
    k_gemm_t<HID><<<(ROWS/128)*(VD/64), 128, 0, stream>>>(h16, wv_h, Cbuf, ROWS, VD);
    k_vcast<<<ROWS*VD/256, 256, 0, stream>>>(Cbuf, VT16);
    // G = h @ WG
    k_gemm_t<HID><<<(ROWS/128)*(VD/64), 128, 0, stream>>>(h16, wg_h, Gf, ROWS, VD);
    // fused retention
    k_attn<<<dim3(SEQL/64, NHEADS, BATCH), 128, 0, stream>>>(Q16, K16, VT16, Ybuf);
    // GroupNorm + SiLU gate -> f16
    k_gngate<<<ROWS*NHEADS/4, 128, 0, stream>>>(Ybuf, Gf, gn_g + L*VD, gn_b + L*VD, a16);
    // ret = gated @ WO ; y = ret + x ; h2 = LN2(y)
    k_gemm_t<VD><<<(ROWS/128)*(HID/64), 128, 0, stream>>>(a16, wo_h, Cbuf, ROWS, HID);
    k_resid_ln<<<ROWS/4, 128, 0, stream>>>(Cbuf, xcur, ln2_g + L*HID, ln2_b + L*HID, ybuf, h16);
    // FFN
    k_gemm_t<HID><<<(ROWS/128)*(FFND/64), 128, 0, stream>>>(h16, w1_h, Cbuf, ROWS, FFND);
    k_gelu<<<ROWS*FFND/256, 256, 0, stream>>>(Cbuf, b1 + L*FFND, a16);
    k_gemm_t<FFND><<<(ROWS/128)*(HID/64), 128, 0, stream>>>(a16, w2_h, Cbuf, ROWS, HID);
    k_addout<<<ROWS*HID/256, 256, 0, stream>>>(Cbuf, b2 + L*HID, ybuf, xout);
  }
}
